// AttentionSequencePoolingLayer_30167850287841
// MI455X (gfx1250) — compile-verified, matrix operand order checked by reading, not run on hardware
//
#include <hip/hip_runtime.h>

typedef _Float16 v8h  __attribute__((ext_vector_type(8)));
typedef _Float16 v16h __attribute__((ext_vector_type(16)));
typedef float    v8f  __attribute__((ext_vector_type(8)));
typedef int      v4i  __attribute__((ext_vector_type(4)));

#define B_   4096
#define T_   200
#define E_   64
#define H1_  80
#define H2_  40
#define K1_  256   // 4*E
#define NTILES 13  // ceil(200/16)
#define TPAD 208   // 13*16
#define H1P  96    // H1 padded to multiple of 32 (K for GEMM2)
#define H2P  48    // H2 padded to multiple of 16

#if defined(__has_builtin)
#if __has_builtin(__builtin_amdgcn_global_load_async_to_lds_b128)
#define HAVE_ASYNC_LDS 1
#endif
#endif

typedef __attribute__((address_space(1))) v4i* gv4i_ptr;
typedef __attribute__((address_space(3))) v4i* lv4i_ptr;

// Load a 16-half WMMA fragment (A or B operand) from LDS.
// Per ISA layout: halves 0..7 at p[off], halves 8..15 at p[off+16].
__device__ __forceinline__ v16h ld_frag(const _Float16* p, int off) {
  union { v16h v; v8h h[2]; } u;
  u.h[0] = *(const v8h*)(p + off);
  u.h[1] = *(const v8h*)(p + off + 16);
  return u.v;
}

__device__ __forceinline__ float sigmoidf(float x) {
  return 1.0f / (1.0f + __expf(-x));
}

__global__ __launch_bounds__(256) void din_attention_kernel(
    const float* __restrict__ q_g, const float* __restrict__ keys_g,
    const int* __restrict__ klen_g,
    const float* __restrict__ W1_g, const float* __restrict__ b1_g,
    const float* __restrict__ W2_g, const float* __restrict__ b2_g,
    const float* __restrict__ W3_g, const float* __restrict__ b3_g,
    float* __restrict__ out_g)
{
  __shared__ float    keysLDS[T_ * E_];     // 51200 B
  __shared__ float    qLDS[E_];             //   256 B
  __shared__ _Float16 W1T[H1_ * K1_];       // 40960 B  [n][k] (transposed, f16)
  __shared__ _Float16 W2T[H2P * H1P];       //  9216 B  [n][k] zero-padded
  __shared__ float    b1LDS[H1_];
  __shared__ float    b2LDS[H2P];
  __shared__ float    w3LDS[H2_];
  __shared__ float    b3LDS;
  __shared__ _Float16 attLDS[16 * K1_];     //  8192 B  one 16-row feature tile
  __shared__ _Float16 h1LDS[16 * H1P];      //  3072 B  (K pad cols stay zero)
  __shared__ _Float16 h2LDS[16 * H2P];      //  1536 B
  __shared__ float    scoresLDS[TPAD];
  __shared__ float    redBuf[256];

  const int tid  = threadIdx.x;
  const int b    = blockIdx.x;
  const int wave = tid >> 5;
  const int lane = tid & 31;

  // ---- Stage keys[b] into LDS -----------------------------------------
  // HBM-bound leg: use the CDNA5 async global->LDS DMA path when available
  // (no VGPR round-trip, tracked by ASYNCcnt, overlaps with the weight
  // transpose below). Fallback: plain float4 copies.
  {
    const float* ksrc = keys_g + (size_t)b * T_ * E_;
#if defined(HAVE_ASYNC_LDS)
    for (int i = tid; i < (T_ * E_) / 4; i += 256) {
      __builtin_amdgcn_global_load_async_to_lds_b128(
          (gv4i_ptr)(ksrc + i * 4),
          (lv4i_ptr)(keysLDS + i * 4),
          0, 0);
    }
#else
    const float4* ks4 = (const float4*)ksrc;
    float4* kdst = (float4*)keysLDS;
    for (int i = tid; i < (T_ * E_) / 4; i += 256) kdst[i] = ks4[i];
#endif
  }

  if (tid < E_)  qLDS[tid]  = q_g[(size_t)b * E_ + tid];
  if (tid < H1_) b1LDS[tid] = b1_g[tid];
  if (tid < H2P) b2LDS[tid] = (tid < H2_) ? b2_g[tid] : 0.0f;
  if (tid < H2_) w3LDS[tid] = W3_g[tid];
  if (tid == 0)  b3LDS = b3_g[0];
  // zero padded regions first (different thread->index mapping than fill)
  for (int i = tid; i < H2P * H1P; i += 256) W2T[i] = (_Float16)0.0f;
  for (int i = tid; i < 16 * H1P; i += 256)  h1LDS[i] = (_Float16)0.0f;
  __syncthreads();
  // transpose + convert weights to f16 (B operands need contiguous-K rows)
  for (int i = tid; i < K1_ * H1_; i += 256) {
    int k = i / H1_, n = i % H1_;
    W1T[n * K1_ + k] = (_Float16)W1_g[i];
  }
  for (int i = tid; i < H1_ * H2_; i += 256) {
    int k = i / H2_, n = i % H2_;
    W2T[n * H1P + k] = (_Float16)W2_g[i];
  }
#if defined(HAVE_ASYNC_LDS)
  // Ensure the async keys DMA has landed in LDS before publishing it.
#if __has_builtin(__builtin_amdgcn_s_wait_asynccnt)
  __builtin_amdgcn_s_wait_asynccnt(0);
#else
  asm volatile("s_wait_asynccnt 0" ::: "memory");
#endif
#endif
  __syncthreads();

  const int len = klen_g[b];

  // ---- Per-16-row tile: features -> MLP (WMMA) -> scores ----------------
  for (int rt = 0; rt < NTILES; ++rt) {
    const int r0 = rt * 16;

    // Build att tile [16 x 256] f16: [q, k, q-k, q*k]
    {
      const int row = tid >> 4;   // 0..15
      const int cg  = tid & 15;   // this thread fills 16 consecutive cols
      const int t   = r0 + row;
      #pragma unroll
      for (int j = 0; j < 16; ++j) {
        const int c = cg * 16 + j;
        const int seg = c >> 6, e = c & 63;
        float v = 0.0f;
        if (t < T_) {
          const float qe = qLDS[e];
          const float ke = keysLDS[t * E_ + e];
          v = (seg == 0) ? qe : (seg == 1) ? ke
            : (seg == 2) ? (qe - ke) : (qe * ke);
        }
        attLDS[row * K1_ + c] = (_Float16)v;
      }
    }
    __syncthreads();

    // GEMM1: [16x256] x [256x80] ; waves 0..4 -> one 16-col tile each
    if (wave < 5) {
      v8f acc = {};
      const int col = wave * 16 + (lane & 15);
      const _Float16* arow = attLDS + (lane & 15) * K1_;
      const _Float16* brow = W1T + col * K1_;
      const int kb = (lane >> 4) * 8;
      #pragma unroll
      for (int kc = 0; kc < 8; ++kc) {
        v16h a  = ld_frag(arow, kc * 32 + kb);
        v16h bb = ld_frag(brow, kc * 32 + kb);
        acc = __builtin_amdgcn_wmma_f32_16x16x32_f16(
            false, a, false, bb, (short)0, acc, false, false);
      }
      const float bias = b1LDS[col];
      const int mofs = (lane & 16) ? 8 : 0;
      #pragma unroll
      for (int r = 0; r < 8; ++r) {
        const float v = sigmoidf(acc[r] + bias);
        h1LDS[(r + mofs) * H1P + col] = (_Float16)v;
      }
    }
    __syncthreads();

    // GEMM2: [16x96] x [96x48] ; waves 0..2
    if (wave < 3) {
      v8f acc = {};
      const int col = wave * 16 + (lane & 15);
      const _Float16* arow = h1LDS + (lane & 15) * H1P;
      const _Float16* brow = W2T + col * H1P;
      const int kb = (lane >> 4) * 8;
      #pragma unroll
      for (int kc = 0; kc < 3; ++kc) {
        v16h a  = ld_frag(arow, kc * 32 + kb);
        v16h bb = ld_frag(brow, kc * 32 + kb);
        acc = __builtin_amdgcn_wmma_f32_16x16x32_f16(
            false, a, false, bb, (short)0, acc, false, false);
      }
      const float bias = b2LDS[col];
      const int mofs = (lane & 16) ? 8 : 0;
      #pragma unroll
      for (int r = 0; r < 8; ++r) {
        const float v = sigmoidf(acc[r] + bias);
        h2LDS[(r + mofs) * H2P + col] = (_Float16)v;
      }
    }
    __syncthreads();

    // GEMM3: [16x40] x [40x1] -> 16 scores (tiny; VALU dot per row)
    if (tid < 16) {
      float acc = b3LDS;
      #pragma unroll
      for (int j = 0; j < H2_; ++j)
        acc += (float)h2LDS[tid * H2P + j] * w3LDS[j];
      scoresLDS[r0 + tid] = acc;
    }
    __syncthreads();
  }

  // ---- Masked softmax over T ------------------------------------------
  const bool valid = (tid < T_) && (tid < len);
  float sc = valid ? scoresLDS[tid] : -3.4e38f;
  redBuf[tid] = sc;
  __syncthreads();
  #pragma unroll
  for (int s = 128; s > 0; s >>= 1) {
    if (tid < s) redBuf[tid] = fmaxf(redBuf[tid], redBuf[tid + s]);
    __syncthreads();
  }
  const float mx = redBuf[0];
  __syncthreads();
  const float ex = valid ? __expf(sc - mx) : 0.0f;
  redBuf[tid] = ex;
  __syncthreads();
  #pragma unroll
  for (int s = 128; s > 0; s >>= 1) {
    if (tid < s) redBuf[tid] += redBuf[tid + s];
    __syncthreads();
  }
  const float inv = 1.0f / redBuf[0];
  __syncthreads();
  if (tid < T_) scoresLDS[tid] = ex * inv;
  __syncthreads();

  // ---- Weighted sum of keys: out[b][e] = sum_t w[t]*keys[t][e] ---------
  {
    const int e    = tid & 63;
    const int part = tid >> 6;          // 4 partial sums of 50 t's each
    const int t0   = part * 50;
    float acc = 0.0f;
    for (int t = t0; t < t0 + 50; ++t)
      acc += scoresLDS[t] * keysLDS[t * E_ + e];
    redBuf[tid] = acc;
    __syncthreads();
    if (tid < 128) redBuf[tid] += redBuf[tid + 128];
    __syncthreads();
    if (tid < 64)
      out_g[(size_t)b * E_ + tid] = redBuf[tid] + redBuf[tid + 64];
  }
}

extern "C" void kernel_launch(void* const* d_in, const int* in_sizes, int n_in,
                              void* d_out, int out_size, void* d_ws, size_t ws_size,
                              hipStream_t stream) {
  const float* q    = (const float*)d_in[0];
  const float* keys = (const float*)d_in[1];
  const int*   kl   = (const int*)d_in[2];
  const float* W1   = (const float*)d_in[3];
  const float* b1   = (const float*)d_in[4];
  const float* W2   = (const float*)d_in[5];
  const float* b2   = (const float*)d_in[6];
  const float* W3   = (const float*)d_in[7];
  const float* b3   = (const float*)d_in[8];
  din_attention_kernel<<<B_, 256, 0, stream>>>(q, keys, kl, W1, b1, W2, b2, W3,
                                               b3, (float*)d_out);
}
